// SinkhornMatcherWithFilters_36180804501550
// MI455X (gfx1250) — matrix operand order repaired and checked by reading, not compiled, not run on hardware
//
#include <hip/hip_runtime.h>
#include <hip/hip_bf16.h>
#include <math.h>

typedef __attribute__((ext_vector_type(16))) __bf16 v16bf;
typedef __attribute__((ext_vector_type(8)))  __bf16 v8bf;
typedef __attribute__((ext_vector_type(8)))  float  v8f;

#define BATCH   8
#define NREAL   2047
#define NPAD    2048
#define DDIM    256
#define TILE_K  32
#define DUSTBIN (-1.0f)   // -UNUSED_SCORE / EPSILON
#define NEG_INF (-3.0e38f)

// ---------------------------------------------------------------------------
// Squared row norms: one wave per row, lane-strided coalesced reads.
// ---------------------------------------------------------------------------
__global__ __launch_bounds__(256)
void norms_kernel(const float* __restrict__ desc, float* __restrict__ out) {
    int wave = threadIdx.x >> 5;
    int lane = threadIdx.x & 31;
    int row  = blockIdx.x * 8 + wave;
    int b    = blockIdx.y;
    if (row >= NREAL) {                 // pad row: define a value, never used
        if (lane == 0 && row < NPAD) out[b * NPAD + row] = 0.0f;
        return;
    }
    const float* p = desc + ((size_t)b * NREAL + row) * DDIM;
    float s = 0.0f;
    #pragma unroll
    for (int d = lane; d < DDIM; d += 32) { float x = p[d]; s += x * x; }
    #pragma unroll
    for (int off = 16; off > 0; off >>= 1) s += __shfl_xor(s, off, 32);
    if (lane == 0) out[b * NPAD + row] = s;
}

// ---------------------------------------------------------------------------
// Cost GEMM via split-bf16 WMMA:  ls = min(2*dot - n1 - n2, 0)
// 128x128 tile per 256-thread block (8 waves), K staged 32-wide in LDS
// as pre-split bf16 hi/lo planes. 3 v_wmma_f32_16x16x32_bf16 per 16x16 tile
// per K-step (hi*hi + hi*lo + lo*hi) ~= f32-accurate GEMM at bf16 rate.
// ---------------------------------------------------------------------------
__global__ __launch_bounds__(256)
void cost_gemm_kernel(const float* __restrict__ d1, const float* __restrict__ d2,
                      const float* __restrict__ n1, const float* __restrict__ n2,
                      float* __restrict__ ls) {
    __shared__ __attribute__((aligned(16))) unsigned short AsH[128 * 32];
    __shared__ __attribute__((aligned(16))) unsigned short AsL[128 * 32];
    __shared__ __attribute__((aligned(16))) unsigned short BsH[128 * 32];
    __shared__ __attribute__((aligned(16))) unsigned short BsL[128 * 32];

    const int b       = blockIdx.z;
    const int rowBase = blockIdx.y * 128;
    const int colBase = blockIdx.x * 128;
    const int tid     = threadIdx.x;
    const int wave    = tid >> 5;
    const int lane    = tid & 31;
    const int wr      = (wave & 1) * 64;   // wave row offset in tile
    const int wc      = (wave >> 1) * 32;  // wave col offset in tile
    const int lhalf   = lane >> 4;         // lane group 0/1
    const int l15     = lane & 15;

    v8f acc[4][2];
    #pragma unroll
    for (int tm = 0; tm < 4; ++tm)
        #pragma unroll
        for (int tn = 0; tn < 2; ++tn)
            #pragma unroll
            for (int i = 0; i < 8; ++i) acc[tm][tn][i] = 0.0f;

    const float* d1b = d1 + (size_t)b * NREAL * DDIM;
    const float* d2b = d2 + (size_t)b * NREAL * DDIM;

    for (int kk = 0; kk < DDIM; kk += TILE_K) {
        // ---- global -> LDS, splitting f32 into bf16 hi/lo planes ----
        #pragma unroll
        for (int j = 0; j < 4; ++j) {
            int idx = tid + j * 256;        // 0..1023
            int r   = idx >> 3;             // 0..127
            int seg = idx & 7;              // 8 x float4 per 32-wide row
            int gr  = rowBase + r; if (gr > NREAL - 1) gr = NREAL - 1;
            int gc  = colBase + r; if (gc > NREAL - 1) gc = NREAL - 1;
            const float4 a  = *(const float4*)(d1b + (size_t)gr * DDIM + kk + seg * 4);
            const float4 bq = *(const float4*)(d2b + (size_t)gc * DDIM + kk + seg * 4);
            const float av[4] = {a.x, a.y, a.z, a.w};
            const float bv[4] = {bq.x, bq.y, bq.z, bq.w};
            #pragma unroll
            for (int e = 0; e < 4; ++e) {
                unsigned int ab = __float_as_uint(av[e]);
                float alo = av[e] - __uint_as_float(ab & 0xFFFF0000u);
                AsH[r * 32 + seg * 4 + e] = (unsigned short)(ab >> 16);
                AsL[r * 32 + seg * 4 + e] = (unsigned short)(__float_as_uint(alo) >> 16);
                unsigned int bb = __float_as_uint(bv[e]);
                float blo = bv[e] - __uint_as_float(bb & 0xFFFF0000u);
                BsH[r * 32 + seg * 4 + e] = (unsigned short)(bb >> 16);
                BsL[r * 32 + seg * 4 + e] = (unsigned short)(__float_as_uint(blo) >> 16);
            }
        }
        // prefetch next K-tile (global_prefetch_b8)
        if (kk + TILE_K < DDIM) {
            int pr = rowBase + (tid & 127); if (pr > NREAL - 1) pr = NREAL - 1;
            int pc = colBase + (tid & 127); if (pc > NREAL - 1) pc = NREAL - 1;
            __builtin_prefetch(d1b + (size_t)pr * DDIM + kk + TILE_K, 0, 3);
            __builtin_prefetch(d2b + (size_t)pc * DDIM + kk + TILE_K, 0, 3);
        }
        __syncthreads();

        // ---- build fragments (ISA 7.12.2 layouts), b128 LDS reads ----
        v16bf aH[4], aL[4], bH[2], bL[2];
        const int akb = lhalf * 8;          // A: K base 0 or 8
        #pragma unroll
        for (int tm = 0; tm < 4; ++tm) {
            int r = wr + tm * 16 + l15;
            const v8bf p0 = *(const v8bf*)&AsH[r * 32 + akb];
            const v8bf p1 = *(const v8bf*)&AsH[r * 32 + 16 + akb];
            const v8bf q0 = *(const v8bf*)&AsL[r * 32 + akb];
            const v8bf q1 = *(const v8bf*)&AsL[r * 32 + 16 + akb];
            #pragma unroll
            for (int i = 0; i < 8; ++i) {
                aH[tm][i] = p0[i]; aH[tm][i + 8] = p1[i];
                aL[tm][i] = q0[i]; aL[tm][i + 8] = q1[i];
            }
        }
        const int bkb = lhalf * 8;          // B: K base 0 or 8
        #pragma unroll
        for (int tn = 0; tn < 2; ++tn) {
            int c = wc + tn * 16 + l15;
            const v8bf p0 = *(const v8bf*)&BsH[c * 32 + bkb];
            const v8bf p1 = *(const v8bf*)&BsH[c * 32 + 16 + bkb];
            const v8bf q0 = *(const v8bf*)&BsL[c * 32 + bkb];
            const v8bf q1 = *(const v8bf*)&BsL[c * 32 + 16 + bkb];
            #pragma unroll
            for (int i = 0; i < 8; ++i) {
                bH[tn][i] = p0[i]; bH[tn][i + 8] = p1[i];
                bL[tn][i] = q0[i]; bL[tn][i + 8] = q1[i];
            }
        }

        // ---- 4x2 output tiles, 3 WMMAs each (hi*hi + hi*lo + lo*hi) ----
        #pragma unroll
        for (int tm = 0; tm < 4; ++tm)
            #pragma unroll
            for (int tn = 0; tn < 2; ++tn) {
                acc[tm][tn] = __builtin_amdgcn_wmma_f32_16x16x32_bf16(
                    false, aH[tm], false, bH[tn], (short)0, acc[tm][tn], false, false);
                acc[tm][tn] = __builtin_amdgcn_wmma_f32_16x16x32_bf16(
                    false, aH[tm], false, bL[tn], (short)0, acc[tm][tn], false, false);
                acc[tm][tn] = __builtin_amdgcn_wmma_f32_16x16x32_bf16(
                    false, aL[tm], false, bH[tn], (short)0, acc[tm][tn], false, false);
            }
        __syncthreads();
    }

    // ---- store log_scores = min(2*dot - n1 - n2, 0) (pads fixed later) ----
    float* lsb = ls + (size_t)b * NPAD * NPAD;
    #pragma unroll
    for (int tm = 0; tm < 4; ++tm) {
        #pragma unroll
        for (int tn = 0; tn < 2; ++tn) {
            int n = colBase + wc + tn * 16 + l15;
            float nn2 = n2[b * NPAD + n];
            #pragma unroll
            for (int i = 0; i < 8; ++i) {
                int m = rowBase + wr + tm * 16 + lhalf * 8 + i;   // C layout
                float val = 2.0f * acc[tm][tn][i] - n1[b * NPAD + m] - nn2;
                lsb[(size_t)m * NPAD + n] = fminf(val, 0.0f);
            }
        }
    }
}

// ---------------------------------------------------------------------------
// Dustbin row/col = -1.0
// ---------------------------------------------------------------------------
__global__ void fill_dustbin_kernel(float* __restrict__ ls) {
    int b = blockIdx.y;
    int i = blockIdx.x * 256 + threadIdx.x;
    float* lsb = ls + (size_t)b * NPAD * NPAD;
    if (i < NPAD) {
        lsb[(size_t)(NPAD - 1) * NPAD + i] = DUSTBIN;
        lsb[(size_t)i * NPAD + (NPAD - 1)] = DUSTBIN;
    }
}

__global__ void init_uv_kernel(float* __restrict__ u, float* __restrict__ v) {
    int i = blockIdx.x * 256 + threadIdx.x;
    if (i < BATCH * NPAD) { u[i] = 0.0f; v[i] = 0.0f; }
}

// ---------------------------------------------------------------------------
// u[b,n] = log_mu[n] - logsumexp_m( ls[b,n,m] + v[b,m] )   (online LSE)
// One block per (b, n); float4 streaming reads (L2-resident matrix).
// ---------------------------------------------------------------------------
__global__ __launch_bounds__(256)
void row_lse_kernel(const float* __restrict__ ls, const float* __restrict__ v,
                    float* __restrict__ u) {
    __shared__ float rm[256], rs[256];
    int b = blockIdx.y, n = blockIdx.x, tid = threadIdx.x;
    const float4* r4 = (const float4*)(ls + ((size_t)b * NPAD + n) * NPAD);
    const float4* v4 = (const float4*)(v + b * NPAD);
    float mx = NEG_INF, s = 0.0f;
    for (int i = tid; i < NPAD / 4; i += 256) {
        float4 a = r4[i], c = v4[i];
        float x[4] = {a.x + c.x, a.y + c.y, a.z + c.z, a.w + c.w};
        #pragma unroll
        for (int e = 0; e < 4; ++e) {
            float xe = x[e];
            if (xe > mx) { s = s * __expf(mx - xe) + 1.0f; mx = xe; }
            else         { s += __expf(xe - mx); }
        }
    }
    rm[tid] = mx; rs[tid] = s;
    for (int off = 128; off > 0; off >>= 1) {
        __syncthreads();
        if (tid < off) {
            float m1 = rm[tid], s1 = rs[tid];
            float m2 = rm[tid + off], s2 = rs[tid + off];
            float mo = fmaxf(m1, m2);
            rs[tid] = s1 * __expf(m1 - mo) + s2 * __expf(m2 - mo);
            rm[tid] = mo;
        }
    }
    if (tid == 0) {
        float lse = rm[0] + __logf(rs[0]);
        float lmu = (n < NREAL) ? 0.0f : __logf((float)NREAL);
        u[b * NPAD + n] = lmu - lse;
    }
}

// ---------------------------------------------------------------------------
// Column LSE, stage 1: per-(32-col, 256-row chunk) partial (max, sum).
// blockDim=256 as 32x8: consecutive lanes read consecutive columns (128B).
// ---------------------------------------------------------------------------
__global__ __launch_bounds__(256)
void col_lse_partial_kernel(const float* __restrict__ ls, const float* __restrict__ u,
                            float* __restrict__ pm, float* __restrict__ ps) {
    __shared__ float rm[8][32], rs[8][32];
    int b = blockIdx.y, chunk = blockIdx.z;
    int tx = threadIdx.x & 31, ty = threadIdx.x >> 5;
    int col = blockIdx.x * 32 + tx;
    const float* lsb = ls + (size_t)b * NPAD * NPAD;
    const float* ub  = u + b * NPAD;
    float mx = NEG_INF, s = 0.0f;
    int n0 = chunk * 256;
    for (int n = n0 + ty; n < n0 + 256; n += 8) {
        float x = lsb[(size_t)n * NPAD + col] + ub[n];
        if (x > mx) { s = s * __expf(mx - x) + 1.0f; mx = x; }
        else        { s += __expf(x - mx); }
    }
    rm[ty][tx] = mx; rs[ty][tx] = s;
    __syncthreads();
    if (ty == 0) {
        #pragma unroll
        for (int y = 1; y < 8; ++y) {
            float m2 = rm[y][tx], s2 = rs[y][tx];
            float mo = fmaxf(mx, m2);
            s = s * __expf(mx - mo) + s2 * __expf(m2 - mo);
            mx = mo;
        }
        size_t o = ((size_t)b * 8 + chunk) * NPAD + col;
        pm[o] = mx; ps[o] = s;
    }
}

// Column LSE, stage 2: combine 8 chunk partials -> v[b,m]
__global__ void col_lse_combine_kernel(const float* __restrict__ pm,
                                       const float* __restrict__ ps,
                                       float* __restrict__ v) {
    int b = blockIdx.y;
    int col = blockIdx.x * 256 + threadIdx.x;
    float mx = NEG_INF, s = 0.0f;
    #pragma unroll
    for (int c = 0; c < 8; ++c) {
        size_t o = ((size_t)b * 8 + c) * NPAD + col;
        float m2 = pm[o], s2 = ps[o];
        float mo = fmaxf(mx, m2);
        s = s * __expf(mx - mo) + s2 * __expf(m2 - mo);
        mx = mo;
    }
    float lnu = (col < NREAL) ? 0.0f : __logf((float)NREAL);
    v[b * NPAD + col] = lnu - (mx + __logf(s));
}

// ---------------------------------------------------------------------------
// P = exp(ls + u + v) in place; per-row top-2 over core cols + dustbin test.
// valid = (best/(second+1e-8) >= 2) && (best - dust >= 0.3)
// ---------------------------------------------------------------------------
__global__ __launch_bounds__(256)
void finalize_kernel(float* __restrict__ P, const float* __restrict__ u,
                     const float* __restrict__ v, float* __restrict__ valid) {
    __shared__ float rb[256], r2[256];
    __shared__ float dustSh;
    int b = blockIdx.y, n = blockIdx.x, tid = threadIdx.x;
    float* row = P + ((size_t)b * NPAD + n) * NPAD;
    const float* vb = v + b * NPAD;
    float un = u[b * NPAD + n];
    float b1 = -1.0f, b2 = -1.0f;
    for (int i = tid; i < NPAD / 4; i += 256) {
        float4 a = *(const float4*)(row + i * 4);
        float4 c = *(const float4*)(vb + i * 4);
        float p[4] = {__expf(a.x + un + c.x), __expf(a.y + un + c.y),
                      __expf(a.z + un + c.z), __expf(a.w + un + c.w)};
        *(float4*)(row + i * 4) = make_float4(p[0], p[1], p[2], p[3]);
        #pragma unroll
        for (int e = 0; e < 4; ++e) {
            int m = i * 4 + e;
            if (m < NREAL) {
                if (p[e] > b1) { b2 = b1; b1 = p[e]; }
                else if (p[e] > b2) { b2 = p[e]; }
            } else if (m == NPAD - 1) {
                dustSh = p[e];          // single writer (tid 255, i = 511)
            }
        }
    }
    rb[tid] = b1; r2[tid] = b2;
    for (int off = 128; off > 0; off >>= 1) {
        __syncthreads();
        if (tid < off) {
            float a1 = rb[tid], a2 = r2[tid];
            float o1 = rb[tid + off], o2 = r2[tid + off];
            rb[tid] = fmaxf(a1, o1);
            r2[tid] = fmaxf(fminf(a1, o1), fmaxf(a2, o2));
        }
    }
    if (tid == 0 && n < NREAL) {
        float best = rb[0], second = r2[0];
        bool ok = (best / (second + 1e-8f) >= 2.0f) && (best - dustSh >= 0.3f);
        valid[b * NREAL + n] = ok ? 1.0f : 0.0f;
    }
}

// ---------------------------------------------------------------------------
extern "C" void kernel_launch(void* const* d_in, const int* in_sizes, int n_in,
                              void* d_out, int out_size, void* d_ws, size_t ws_size,
                              hipStream_t stream) {
    const float* desc1 = (const float*)d_in[0];
    const float* desc2 = (const float*)d_in[1];

    float* P     = (float*)d_out;                       // ls, then P in place
    float* valid = P + (size_t)BATCH * NPAD * NPAD;

    float* ws = (float*)d_ws;
    float* n1 = ws;
    float* n2 = n1 + BATCH * NPAD;
    float* u  = n2 + BATCH * NPAD;
    float* v  = u  + BATCH * NPAD;
    float* pm = v  + BATCH * NPAD;
    float* ps = pm + (size_t)BATCH * 8 * NPAD;

    dim3 blk(256);
    norms_kernel<<<dim3(NPAD / 8, BATCH), blk, 0, stream>>>(desc1, n1);
    norms_kernel<<<dim3(NPAD / 8, BATCH), blk, 0, stream>>>(desc2, n2);
    cost_gemm_kernel<<<dim3(NPAD / 128, NPAD / 128, BATCH), blk, 0, stream>>>(
        desc1, desc2, n1, n2, P);
    fill_dustbin_kernel<<<dim3(NPAD / 256, BATCH), blk, 0, stream>>>(P);
    init_uv_kernel<<<dim3((BATCH * NPAD) / 256), blk, 0, stream>>>(u, v);

    for (int it = 0; it < 20; ++it) {
        row_lse_kernel<<<dim3(NPAD, BATCH), blk, 0, stream>>>(P, v, u);
        col_lse_partial_kernel<<<dim3(NPAD / 32, BATCH, 8), blk, 0, stream>>>(P, u, pm, ps);
        col_lse_combine_kernel<<<dim3(NPAD / 256, BATCH), blk, 0, stream>>>(pm, ps, v);
    }

    finalize_kernel<<<dim3(NPAD, BATCH), blk, 0, stream>>>(P, u, v, valid);
}
